// MoE_Block_24068996727023
// MI455X (gfx1250) — matrix operand order repaired and checked by reading, not compiled, not run on hardware
//
#include <hip/hip_runtime.h>
#include <cstdint>
#include <cstddef>

// ---------------------------------------------------------------------------
// Problem constants (reference: B=8, N=1024, D=768, H=12, DH=64, E=5, HID=1344)
// ---------------------------------------------------------------------------
#define TOK   8192          // B*N tokens
#define DIM   768
#define D3    2304          // 3*D
#define NH    12
#define DHEAD 64
#define NEXP  5
#define HIDN  1344

typedef __attribute__((ext_vector_type(8)))  float  v8f;
typedef __attribute__((ext_vector_type(16))) __bf16 v16bf;
typedef __attribute__((ext_vector_type(8)))  __bf16 v8bf;

// Build a 16x32 bf16 A-fragment for one lane from row-major LDS:
// lane holds row m=lane&15; contiguous K=[hf*16, hf*16+16).
static __device__ __forceinline__ v16bf make_afrag(const __bf16* p) {
  v8bf lo = *(const v8bf*)p;
  v8bf hi = *(const v8bf*)(p + 8);
  return __builtin_shufflevector(lo, hi, 0,1,2,3,4,5,6,7,8,9,10,11,12,13,14,15);
}

static __device__ __forceinline__ v8f vzero8() {
  v8f z = {0.f,0.f,0.f,0.f,0.f,0.f,0.f,0.f};
  return z;
}

// Async global->LDS 16B copy (GLOBAL_LOAD_ASYNC_TO_LDS_B128, ASYNCcnt).
// lds_off: LDS byte offset (low 32 bits of generic shared address, per ISA
// flat->LDS truncation). gaddr: 64-bit global address (VGPR pair).
static __device__ __forceinline__ void async_copy16(unsigned lds_off,
                                                    unsigned long long gaddr) {
  asm volatile("global_load_async_to_lds_b128 %0, %1, off"
               :: "v"(lds_off), "v"(gaddr) : "memory");
}
static __device__ __forceinline__ void async_wait0() {
  asm volatile("s_wait_asynccnt 0x0" ::: "memory");
}

// ---------------------------------------------------------------------------
// fp32 -> bf16 weight conversion (grid-stride)
// ---------------------------------------------------------------------------
__global__ __launch_bounds__(256) void cvt_bf16(const float* __restrict__ s,
                                                __bf16* __restrict__ d, int n) {
  for (int i = blockIdx.x * blockDim.x + threadIdx.x; i < n;
       i += gridDim.x * blockDim.x)
    d[i] = (__bf16)s[i];
}

// ---------------------------------------------------------------------------
// LayerNorm over D=768, one block (256 thr) per token, bf16 output
// ---------------------------------------------------------------------------
__global__ __launch_bounds__(256) void ln_kernel(const float* __restrict__ x,
                                                 const float* __restrict__ g,
                                                 const float* __restrict__ bta,
                                                 __bf16* __restrict__ out) {
  __shared__ float red[256];
  const int t   = blockIdx.x;
  const int tid = threadIdx.x;
  const float* row = x + (size_t)t * DIM;
  float v0 = row[tid], v1 = row[tid + 256], v2 = row[tid + 512];

  red[tid] = v0 + v1 + v2;
  __syncthreads();
  for (int off = 128; off > 0; off >>= 1) {
    if (tid < off) red[tid] += red[tid + off];
    __syncthreads();
  }
  const float mu = red[0] * (1.f / (float)DIM);
  __syncthreads();

  float d0 = v0 - mu, d1 = v1 - mu, d2 = v2 - mu;
  red[tid] = d0 * d0 + d1 * d1 + d2 * d2;
  __syncthreads();
  for (int off = 128; off > 0; off >>= 1) {
    if (tid < off) red[tid] += red[tid + off];
    __syncthreads();
  }
  const float rs = rsqrtf(red[0] * (1.f / (float)DIM) + 1e-5f);

  __bf16* orow = out + (size_t)t * DIM;
  orow[tid]       = (__bf16)(d0 * rs * g[tid]       + bta[tid]);
  orow[tid + 256] = (__bf16)(d1 * rs * g[tid + 256] + bta[tid + 256]);
  orow[tid + 512] = (__bf16)(d2 * rs * g[tid + 512] + bta[tid + 512]);
}

// ---------------------------------------------------------------------------
// MoE gates: softmax(xn @ centroids^T) per token, E=5 (tiny -> VALU fp32)
// ---------------------------------------------------------------------------
__global__ __launch_bounds__(256) void gates_kernel(const __bf16* __restrict__ xn,
                                                    const float* __restrict__ cent,
                                                    float* __restrict__ gates) {
  __shared__ float red[NEXP][256];
  const int t = blockIdx.x, tid = threadIdx.x;
  float p[NEXP] = {0.f, 0.f, 0.f, 0.f, 0.f};
  for (int i = 0; i < 3; ++i) {
    int d = tid + i * 256;
    float v = (float)xn[(size_t)t * DIM + d];
    #pragma unroll
    for (int e = 0; e < NEXP; ++e) p[e] += v * cent[e * DIM + d];
  }
  #pragma unroll
  for (int e = 0; e < NEXP; ++e) red[e][tid] = p[e];
  __syncthreads();
  for (int off = 128; off > 0; off >>= 1) {
    if (tid < off)
      #pragma unroll
      for (int e = 0; e < NEXP; ++e) red[e][tid] += red[e][tid + off];
    __syncthreads();
  }
  if (tid == 0) {
    float mx = red[0][0];
    #pragma unroll
    for (int e = 1; e < NEXP; ++e) mx = fmaxf(mx, red[e][0]);
    float ex[NEXP], s = 0.f;
    #pragma unroll
    for (int e = 0; e < NEXP; ++e) { ex[e] = __expf(red[e][0] - mx); s += ex[e]; }
    const float inv = 1.f / s;
    #pragma unroll
    for (int e = 0; e < NEXP; ++e) gates[(size_t)t * NEXP + e] = ex[e] * inv;
  }
}

// ---------------------------------------------------------------------------
// Generic bf16 WMMA GEMM: C(MxN) = A(MxK) @ B(KxN), fp32 accumulate.
// Tile 128x64, 256 thr = 8 waves (4 M x 2 N), wave tile 32x32 = 2x2 WMMA.
// A tile staged via async global->LDS engine; B tile transposed via ds_store.
// ---------------------------------------------------------------------------
enum { EPI_BF16 = 0, EPI_BIAS_RES = 1, EPI_BIAS_GELU = 2, EPI_GATE_ACC = 3 };

template <int EPI>
__global__ __launch_bounds__(256)
void gemm_bf16(const __bf16* __restrict__ A, const __bf16* __restrict__ Bm,
               const float* __restrict__ bias, const float* __restrict__ res,
               const float* __restrict__ gates, int expert,
               void* __restrict__ outp, int M, int N, int K) {
  __shared__ __bf16 As[128][32];   // A tile, row-major
  __shared__ __bf16 Bt[64][32];    // B tile, transposed to [n][k]
  const int tid  = threadIdx.x;
  const int wave = tid >> 5, lane = tid & 31;
  const int hf   = lane >> 4, l16 = lane & 15;
  const int wm   = (wave & 3) * 32;
  const int wn   = (wave >> 2) * 32;
  const int m0   = blockIdx.x * 128;
  const int n0   = blockIdx.y * 64;

  v8f acc[2][2] = {{vzero8(), vzero8()}, {vzero8(), vzero8()}};

  const int arow  = tid >> 1;
  const int ahalf = (tid & 1) * 16;

  for (int k0 = 0; k0 < K; k0 += 32) {
    // stage A: 128x32 bf16, 32B per thread via async global->LDS (2x B128)
    {
      const __bf16* src = A + (size_t)(m0 + arow) * K + k0 + ahalf;
      const unsigned lds = (unsigned)(uintptr_t)&As[arow][ahalf];
      const unsigned long long ga = (unsigned long long)(uintptr_t)src;
      async_copy16(lds, ga);
      async_copy16(lds + 16u, ga + 16ull);
      if (k0 + 32 < K) __builtin_prefetch(src + 32, 0, 1);
    }
    // stage B transposed: Bt[n][k] = B[k0+k][n0+n]
    #pragma unroll
    for (int i = 0; i < 8; ++i) {
      int idx = tid + i * 256;            // 0..2047
      int n = idx & 63, kk = idx >> 6;
      Bt[n][kk] = Bm[(size_t)(k0 + kk) * N + (n0 + n)];
    }
    async_wait0();
    __syncthreads();

    v16bf af[2], bfv[2];
    #pragma unroll
    for (int t = 0; t < 2; ++t)
      af[t] = make_afrag(&As[wm + t * 16 + l16][hf * 16]);
    #pragma unroll
    for (int t = 0; t < 2; ++t)
      bfv[t] = *(const v16bf*)&Bt[wn + t * 16 + l16][hf * 16];

    #pragma unroll
    for (int i = 0; i < 2; ++i)
      #pragma unroll
      for (int j = 0; j < 2; ++j)
        acc[i][j] = __builtin_amdgcn_wmma_f32_16x16x32_bf16(
            false, af[i], false, bfv[j], (short)0, acc[i][j], false, false);
    __syncthreads();
  }

  // epilogue: C layout -> element (m,n): vgpr r = m&7, lane = (m>=8)*16 + n
  #pragma unroll
  for (int i = 0; i < 2; ++i) {
    #pragma unroll
    for (int j = 0; j < 2; ++j) {
      #pragma unroll
      for (int r = 0; r < 8; ++r) {
        const int m = m0 + wm + i * 16 + r + hf * 8;
        const int n = n0 + wn + j * 16 + l16;
        const size_t o = (size_t)m * N + n;
        const float v = acc[i][j][r];
        if constexpr (EPI == EPI_BF16) {
          ((__bf16*)outp)[o] = (__bf16)v;
        } else if constexpr (EPI == EPI_BIAS_RES) {
          ((float*)outp)[o] = v + bias[n] + res[o];
        } else if constexpr (EPI == EPI_BIAS_GELU) {
          const float t = v + bias[n];
          ((__bf16*)outp)[o] =
              (__bf16)(0.5f * t * (1.f + erff(t * 0.70710678118654752f)));
        } else {  // EPI_GATE_ACC
          const float g = gates[(size_t)m * NEXP + expert];
          ((float*)outp)[o] += g * (v + bias[n]);
        }
      }
    }
  }
}

// ---------------------------------------------------------------------------
// Attention: one block per (b,h, 64-query tile). 128 thr = 4 waves.
// Full 64x1024 score block lives in LDS (279KB dynamic, <320KB WGP LDS).
// S = (Q*0.125) @ K^T via WMMA; fp32 softmax; O = P @ V via WMMA.
// K tiles staged via async global->LDS engine.
// ---------------------------------------------------------------------------
#define ATTN_LDS (64 * 1024 * 4 + 64 * 64 * 2 * 2 + 64 * 4)

__global__ __launch_bounds__(128)
void attn_kernel(const __bf16* __restrict__ qkv, __bf16* __restrict__ aout) {
  extern __shared__ char smem[];
  float*  S      = (float*)smem;                 // [64][1024] fp32
  __bf16* Qs     = (__bf16*)(smem + 64 * 1024 * 4);  // [64][64]
  __bf16* KV     = Qs + 64 * 64;                 // [64][64] (K, then V^T)
  float*  rowscl = (float*)(KV + 64 * 64);       // [64]

  const int bh = blockIdx.x;
  const int b = bh / NH, h = bh % NH;
  const int rt  = blockIdx.y;                    // query tile (64 rows)
  const int tid = threadIdx.x;
  const int wave = tid >> 5, lane = tid & 31;
  const int hf = lane >> 4, l16 = lane & 15;
  const size_t tok0 = (size_t)b * 1024;

  // stage Q, pre-scaled by DH^-0.5 = 0.125 (exact in bf16)
  for (int i = 0; i < 32; ++i) {
    int idx = tid + i * 128;
    int m = idx >> 6, d = idx & 63;
    float q = (float)qkv[(tok0 + rt * 64 + m) * D3 + h * DHEAD + d];
    Qs[m * 64 + d] = (__bf16)(q * 0.125f);
  }
  __syncthreads();

  // S = Qs @ K^T ; K tile staged row-major == transposed-B layout (async copy)
  for (int kt = 0; kt < 16; ++kt) {
    #pragma unroll
    for (int i = 0; i < 4; ++i) {
      int c = tid + i * 128;                    // 16B chunk id, 0..511
      int key = c >> 3, d8 = (c & 7) * 8;
      const __bf16* src =
          qkv + (tok0 + kt * 64 + key) * D3 + DIM + h * DHEAD + d8;
      async_copy16((unsigned)(uintptr_t)&KV[key * 64 + d8],
                   (unsigned long long)(uintptr_t)src);
    }
    async_wait0();
    __syncthreads();
    #pragma unroll
    for (int tn = 0; tn < 4; ++tn) {
      v8f acc = vzero8();
      #pragma unroll
      for (int ks = 0; ks < 2; ++ks) {
        v16bf a = make_afrag(&Qs[(wave * 16 + l16) * 64 + ks * 32 + hf * 16]);
        v16bf bb =
            *(const v16bf*)&KV[(tn * 16 + l16) * 64 + ks * 32 + hf * 16];
        acc = __builtin_amdgcn_wmma_f32_16x16x32_bf16(
            false, a, false, bb, (short)0, acc, false, false);
      }
      #pragma unroll
      for (int r = 0; r < 8; ++r)
        S[(wave * 16 + r + hf * 8) * 1024 + kt * 64 + tn * 16 + l16] = acc[r];
    }
    __syncthreads();
  }

  // row softmax (one thread per row); 1/sum folded into P fragments later
  if (tid < 64) {
    float* Sr = S + tid * 1024;
    float mx = -3.4e38f;
    for (int j = 0; j < 1024; ++j) mx = fmaxf(mx, Sr[j]);
    float sum = 0.f;
    for (int j = 0; j < 1024; ++j) {
      float e = __expf(Sr[j] - mx);
      Sr[j] = e;
      sum += e;
    }
    rowscl[tid] = 1.f / sum;
  }
  __syncthreads();

  // O = P @ V ; V staged transposed -> Vt[d][key] (manual, reshape needed)
  v8f oacc[4] = {vzero8(), vzero8(), vzero8(), vzero8()};
  for (int kt = 0; kt < 16; ++kt) {
    for (int i = 0; i < 32; ++i) {
      int idx = tid + i * 128;
      int d = idx & 63, key = idx >> 6;
      KV[d * 64 + key] =
          qkv[(tok0 + kt * 64 + key) * D3 + 2 * DIM + h * DHEAD + d];
    }
    __syncthreads();
    const float scl = rowscl[wave * 16 + l16];
    const float* Sp = S + (wave * 16 + l16) * 1024 + kt * 64;
    #pragma unroll
    for (int ks = 0; ks < 2; ++ks) {
      v16bf a;
      const float* p0 = Sp + ks * 32 + hf * 16;
      #pragma unroll
      for (int q = 0; q < 8; ++q) {
        a[q]     = (__bf16)(p0[q] * scl);
        a[q + 8] = (__bf16)(p0[q + 8] * scl);
      }
      #pragma unroll
      for (int tn = 0; tn < 4; ++tn) {
        v16bf bb =
            *(const v16bf*)&KV[(tn * 16 + l16) * 64 + ks * 32 + hf * 16];
        oacc[tn] = __builtin_amdgcn_wmma_f32_16x16x32_bf16(
            false, a, false, bb, (short)0, oacc[tn], false, false);
      }
    }
    __syncthreads();
  }

  // store attention output, bf16 [token][h*64 + d]
  #pragma unroll
  for (int tn = 0; tn < 4; ++tn)
    #pragma unroll
    for (int r = 0; r < 8; ++r) {
      const size_t tok = tok0 + rt * 64 + wave * 16 + r + hf * 8;
      const int d = h * DHEAD + tn * 16 + l16;
      aout[tok * DIM + d] = (__bf16)oacc[tn][r];
    }
}

// ---------------------------------------------------------------------------
// Host orchestration
// ---------------------------------------------------------------------------
extern "C" void kernel_launch(void* const* d_in, const int* in_sizes, int n_in,
                              void* d_out, int out_size, void* d_ws,
                              size_t ws_size, hipStream_t stream) {
  (void)in_sizes; (void)n_in; (void)out_size; (void)ws_size;

  const float* x      = (const float*)d_in[0];
  const float* ln1_g  = (const float*)d_in[1];
  const float* ln1_b  = (const float*)d_in[2];
  const float* qkv_w  = (const float*)d_in[3];
  const float* proj_w = (const float*)d_in[4];
  const float* proj_b = (const float*)d_in[5];
  const float* ln2_g  = (const float*)d_in[6];
  const float* ln2_b  = (const float*)d_in[7];
  const float* cent   = (const float*)d_in[8];
  const float* w1     = (const float*)d_in[9];
  const float* b1     = (const float*)d_in[10];
  const float* w2     = (const float*)d_in[11];
  const float* b2     = (const float*)d_in[12];
  float* out = (float*)d_out;  // holds x1 after proj, final result after MoE

  // workspace partition (~110 MB)
  char* ws = (char*)d_ws;
  auto alloc = [&](size_t bytes) {
    char* p = ws;
    ws += (bytes + 255) & ~(size_t)255;
    return p;
  };
  __bf16* qkvw_bf = (__bf16*)alloc((size_t)DIM * D3 * 2);
  __bf16* projw_bf = (__bf16*)alloc((size_t)DIM * DIM * 2);
  __bf16* w1_bf = (__bf16*)alloc((size_t)NEXP * DIM * HIDN * 2);
  __bf16* w2_bf = (__bf16*)alloc((size_t)NEXP * HIDN * DIM * 2);
  __bf16* xn = (__bf16*)alloc((size_t)TOK * DIM * 2);      // LN1 then LN2 out
  __bf16* qkv_bf = (__bf16*)alloc((size_t)TOK * D3 * 2);
  __bf16* attn_bf = (__bf16*)alloc((size_t)TOK * DIM * 2);
  float* gates = (float*)alloc((size_t)TOK * NEXP * 4);
  __bf16* Hbuf = (__bf16*)alloc((size_t)TOK * HIDN * 2);

  // 1) weights -> bf16
  cvt_bf16<<<1024, 256, 0, stream>>>(qkv_w, qkvw_bf, DIM * D3);
  cvt_bf16<<<1024, 256, 0, stream>>>(proj_w, projw_bf, DIM * DIM);
  cvt_bf16<<<1024, 256, 0, stream>>>(w1, w1_bf, NEXP * DIM * HIDN);
  cvt_bf16<<<1024, 256, 0, stream>>>(w2, w2_bf, NEXP * HIDN * DIM);

  // 2) LN1
  ln_kernel<<<TOK, 256, 0, stream>>>(x, ln1_g, ln1_b, xn);

  // 3) QKV = xn @ qkv_w   [8192 x 2304]
  gemm_bf16<EPI_BF16><<<dim3(TOK / 128, D3 / 64), 256, 0, stream>>>(
      xn, qkvw_bf, nullptr, nullptr, nullptr, 0, qkv_bf, TOK, D3, DIM);

  // 4) attention (279KB dynamic LDS per workgroup)
  hipFuncSetAttribute(reinterpret_cast<const void*>(attn_kernel),
                      hipFuncAttributeMaxDynamicSharedMemorySize,
                      (int)ATTN_LDS);
  attn_kernel<<<dim3(8 * NH, 1024 / 64), 128, ATTN_LDS, stream>>>(qkv_bf,
                                                                  attn_bf);

  // 5) x1 = x + attn @ proj_w + proj_b  -> d_out (fp32)
  gemm_bf16<EPI_BIAS_RES><<<dim3(TOK / 128, DIM / 64), 256, 0, stream>>>(
      attn_bf, projw_bf, proj_b, x, nullptr, 0, out, TOK, DIM, DIM);

  // 6) LN2 on x1
  ln_kernel<<<TOK, 256, 0, stream>>>(out, ln2_g, ln2_b, xn);

  // 7) gates = softmax(xn @ centroids^T)
  gates_kernel<<<TOK, 256, 0, stream>>>(xn, cent, gates);

  // 8) dense MoE, experts sequential:  out += g_e * (gelu(xn@w1_e+b1_e)@w2_e+b2_e)
  for (int e = 0; e < NEXP; ++e) {
    gemm_bf16<EPI_BIAS_GELU><<<dim3(TOK / 128, HIDN / 64), 256, 0, stream>>>(
        xn, w1_bf + (size_t)e * DIM * HIDN, b1 + (size_t)e * HIDN, nullptr,
        nullptr, 0, Hbuf, TOK, HIDN, DIM);
    gemm_bf16<EPI_GATE_ACC><<<dim3(TOK / 128, DIM / 64), 256, 0, stream>>>(
        Hbuf, w2_bf + (size_t)e * HIDN * DIM, b2 + (size_t)e * DIM, nullptr,
        gates, e, out, TOK, DIM, HIDN);
  }
}
